// SNN_38491496907325
// MI455X (gfx1250) — matrix operand order repaired and checked by reading, not compiled, not run on hardware
//
#include <hip/hip_runtime.h>

// SNN: phase A = batched receptive-field GEMM (bf16 WMMA, K-split partials),
//      phase A2 = partial reduction, phase B = sequential scan.
//
// Phase A: C[mat][t][m] = rf[mat] @ stim[mat]^T, M=1200 K=76800 N=100(T).
//   One wave owns (ksplit, mat, mtile) and all 7 N-tiles: A fragment loaded
//   once per k-step, 7 WMMAs issued per step. f32->bf16 via single v_perm_b32.
// Phase B: single 1024-thread workgroup runs the T=100 scan with bit-packed
//   spike-delay buffers in LDS and wave-per-post synaptic reductions.

#define P_IN    76800
#define T_STEPS 100
#define NSA     1200
#define NCN1    300
#define NCN     150
#define VTHR    30.0f
#define PSPDEC  0.9048374180359595f   // exp(-1/10)

#define KSPLIT  8
#define KCHUNK  (P_IN / KSPLIT)       // 9600
#define NT      7                     // ceil(100/16)
#define MT      (NSA / 16)            // 75
#define SLAB    (2 * T_STEPS * NSA)   // one partial slab: 240000 floats

typedef __attribute__((ext_vector_type(16))) __bf16   v16bf;
typedef __attribute__((ext_vector_type(8)))  float    v8f;
typedef __attribute__((ext_vector_type(8)))  unsigned v8u;

union ABFrag { v8u u; v16bf b; };

// Truncating f32 pair -> packed bf16x2: one v_perm_b32.
__device__ __forceinline__ unsigned pk_bf16(float lo, float hi) {
  return __builtin_amdgcn_perm(__float_as_uint(hi), __float_as_uint(lo),
                               0x07060302u);
}

// ---------------------------------------------------------------------------
// Phase A: rf GEMM with K-split partials.
// A-frag layout (16-bit A 16x32): lane<16 row=lane K={0..7,16..23};
//                                 lane>=16 row=lane-16 K={8..15,24..31}.
// B-frag layout (16-bit B 32x16): lanes 0-15 col N=lane hold K={0..7,16..23};
//                                 lanes 16-31 col N=lane-16 hold K={8..15,24..31}.
// C/D layout: VGPR r -> M = r + 8*(lane>>4), N = lane&15.
// ---------------------------------------------------------------------------
__global__ __launch_bounds__(256) void snn_rf_gemm(
    const float* __restrict__ stim,       // (T, P)
    const float* __restrict__ diff_stim,  // (T, P)
    const float* __restrict__ sa_rf,      // (NSA, P)
    const float* __restrict__ ra_rf,      // (NSA, P)
    float* __restrict__ part)             // [KSPLIT][2][T][NSA]
{
  const int wave = threadIdx.x >> 5;
  const int lane = threadIdx.x & 31;
  const int wg   = blockIdx.x * 8 + wave;        // 0..1199
  if (wg >= KSPLIT * 2 * MT) return;

  const int ks    = wg / (2 * MT);
  const int rem   = wg % (2 * MT);
  const int mat   = rem / MT;
  const int mtile = rem % MT;

  const float* __restrict__ A = mat ? ra_rf : sa_rf;
  const float* __restrict__ B = mat ? diff_stim : stim;
  float* __restrict__ C = part + (size_t)ks * SLAB + (size_t)mat * T_STEPS * NSA;

  const int row = lane & 15;
  const int hi  = lane >> 4;

  const float* __restrict__ arow = A + (size_t)(mtile * 16 + row) * P_IN;

  bool tval[NT];
  const float* bptr[NT];
  #pragma unroll
  for (int nt = 0; nt < NT; ++nt) {
    const int t = nt * 16 + row;                 // B column (time) for lane
    tval[nt] = (t < T_STEPS);
    bptr[nt] = B + (size_t)(tval[nt] ? t : 0) * P_IN;
  }

  v8f acc[NT];
  #pragma unroll
  for (int nt = 0; nt < NT; ++nt) acc[nt] = (v8f){};

  const int k0 = ks * KCHUNK;
  const int k1 = k0 + KCHUNK;

  for (int k = k0; k < k1; k += 32) {
    __builtin_prefetch(arow + k + 2048, 0, 0);   // global_prefetch_b8

    // ---- A fragment: two 8-float runs per lane, one load+pack per k-step ----
    ABFrag af;
    {
      const float4 q0 = *(const float4*)(arow + k + hi * 8);
      const float4 q1 = *(const float4*)(arow + k + hi * 8 + 4);
      const float4 q2 = *(const float4*)(arow + k + 16 + hi * 8);
      const float4 q3 = *(const float4*)(arow + k + 16 + hi * 8 + 4);
      af.u[0] = pk_bf16(q0.x, q0.y); af.u[1] = pk_bf16(q0.z, q0.w);
      af.u[2] = pk_bf16(q1.x, q1.y); af.u[3] = pk_bf16(q1.z, q1.w);
      af.u[4] = pk_bf16(q2.x, q2.y); af.u[5] = pk_bf16(q2.z, q2.w);
      af.u[6] = pk_bf16(q3.x, q3.y); af.u[7] = pk_bf16(q3.z, q3.w);
    }

    // ---- 7 N-tiles share the A fragment: 7 WMMAs per k-step ----
    #pragma unroll
    for (int nt = 0; nt < NT; ++nt) {
      const float* bp = bptr[nt];
      ABFrag bfr;
      const float4 r0 = *(const float4*)(bp + k + hi * 8);
      const float4 r1 = *(const float4*)(bp + k + hi * 8 + 4);
      const float4 r2 = *(const float4*)(bp + k + 16 + hi * 8);
      const float4 r3 = *(const float4*)(bp + k + 16 + hi * 8 + 4);
      bfr.u[0] = pk_bf16(r0.x, r0.y); bfr.u[1] = pk_bf16(r0.z, r0.w);
      bfr.u[2] = pk_bf16(r1.x, r1.y); bfr.u[3] = pk_bf16(r1.z, r1.w);
      bfr.u[4] = pk_bf16(r2.x, r2.y); bfr.u[5] = pk_bf16(r2.z, r2.w);
      bfr.u[6] = pk_bf16(r3.x, r3.y); bfr.u[7] = pk_bf16(r3.z, r3.w);
      acc[nt] = __builtin_amdgcn_wmma_f32_16x16x32_bf16(
          false, af.b, false, bfr.b, (short)0, acc[nt], false, false);
    }
  }

  #pragma unroll
  for (int nt = 0; nt < NT; ++nt) {
    if (tval[nt]) {
      const int t = nt * 16 + row;
      #pragma unroll
      for (int r = 0; r < 8; ++r) {
        const int m = mtile * 16 + 8 * hi + r;
        C[(size_t)t * NSA + m] = acc[nt][r];
      }
    }
  }
}

// ---------------------------------------------------------------------------
// Phase A2: sum the KSPLIT partial slabs into slab 0 (deterministic).
// ---------------------------------------------------------------------------
__global__ __launch_bounds__(256) void snn_reduce(float* __restrict__ part) {
  const int idx = blockIdx.x * 256 + threadIdx.x;
  if (idx >= SLAB) return;
  float s = 0.0f;
  #pragma unroll
  for (int ks = 0; ks < KSPLIT; ++ks) s += part[(size_t)ks * SLAB + idx];
  part[idx] = s;
}

// ---------------------------------------------------------------------------
// Phase B: sequential scan in one workgroup.
// ---------------------------------------------------------------------------
__device__ __forceinline__ float izh_step(float& v, float& u, float I,
                                          float a, float b, float c, float d) {
  v = v + (0.04f * v * v + 5.0f * v + 140.0f - u + I);
  u = u + a * (b * v - u);
  float s = (v >= VTHR) ? 1.0f : 0.0f;
  if (s > 0.0f) { v = c; u = u + d; }
  return s;
}

__device__ __forceinline__ float wave_reduce(float x) {
  #pragma unroll
  for (int off = 16; off > 0; off >>= 1) x += __shfl_xor(x, off, 32);
  return x;
}

// out[post] = sum_pre w[post,pre] * ((buf[pre] >> dl[post,pre]) & 1)
// (optionally two weight matrices sharing the delay table)
__device__ __forceinline__ void syn_dual(
    const float* __restrict__ w0, const float* __restrict__ w1,
    const int* __restrict__ dl, const unsigned* __restrict__ buf,
    int npost, int npre, float* __restrict__ out0, float* __restrict__ out1,
    int wave, int lane, int nwaves)
{
  for (int post = wave; post < npost; post += nwaves) {
    const float* r0 = w0 + (size_t)post * npre;
    const float* r1 = w1 ? (w1 + (size_t)post * npre) : nullptr;
    const int*   rd = dl + (size_t)post * npre;
    float a0 = 0.0f, a1 = 0.0f;
    for (int pre = lane; pre < npre; pre += 32) {
      const float sp = (float)((buf[pre] >> rd[pre]) & 1u);
      a0 += r0[pre] * sp;
      if (r1) a1 += r1[pre] * sp;
    }
    a0 = wave_reduce(a0);
    if (r1) a1 = wave_reduce(a1);
    if (lane == 0) { out0[post] = a0; if (r1) out1[post] = a1; }
  }
}

__global__ __launch_bounds__(1024) void snn_scan(
    const float* __restrict__ I_ff,          // [2][T][NSA] (reduced slab 0)
    const float* __restrict__ sa_cn_in_w, const float* __restrict__ sa_cn_pn_w,
    const float* __restrict__ sa_intopn_w,
    const float* __restrict__ ra_cn_in_w, const float* __restrict__ ra_cn_pn_w,
    const float* __restrict__ ra_intopn_w,
    const float* __restrict__ cn_in_sa_w, const float* __restrict__ cn_pn_sa_w,
    const float* __restrict__ cn_in_ra_w, const float* __restrict__ cn_pn_ra_w,
    const float* __restrict__ cn_intopn_w,
    const int* __restrict__ sa_cn_SD, const int* __restrict__ sa_intopn_DN,
    const int* __restrict__ ra_cn_SD, const int* __restrict__ ra_intopn_DN,
    const int* __restrict__ cn_sa_SD, const int* __restrict__ cn_ra_SD,
    const int* __restrict__ cn_intopn_DN,
    float* __restrict__ out)                 // [2][T][NSA]
{
  __shared__ float v_sa0[NSA], u_sa0[NSA];  __shared__ unsigned b_sa0[NSA];
  __shared__ float v_ra0[NSA], u_ra0[NSA];  __shared__ unsigned b_ra0[NSA];
  __shared__ float v_sa1[NCN1], u_sa1[NCN1]; __shared__ unsigned b_sa1[NCN1];
  __shared__ float v_sa2[NCN1], u_sa2[NCN1]; __shared__ unsigned b_sa2[NCN1];
  __shared__ float v_ra1[NCN1], u_ra1[NCN1]; __shared__ unsigned b_ra1[NCN1];
  __shared__ float v_ra2[NCN1], u_ra2[NCN1]; __shared__ unsigned b_ra2[NCN1];
  __shared__ float v_cn0[NCN],  u_cn0[NCN];  __shared__ unsigned b_cn0[NCN];
  __shared__ float v_cn1[NCN],  u_cn1[NCN];
  __shared__ float psp_sa[NCN1], psp_ra[NCN1], psp_cn[NCN];
  __shared__ float I_sa1[NCN1], pn1[NCN1], I_ra1[NCN1], rpn1[NCN1];
  __shared__ float cnA[NCN], cnB[NCN], cnC[NCN], cnD[NCN];

  const int tid   = threadIdx.x;
  const int lane  = tid & 31;
  const int wave  = tid >> 5;
  const int nwav  = blockDim.x >> 5;        // 32

  // init
  for (int n = tid; n < NSA; n += blockDim.x) {
    v_sa0[n] = -65.0f; u_sa0[n] = -13.0f; b_sa0[n] = 0u;
    v_ra0[n] = -65.0f; u_ra0[n] = -13.0f; b_ra0[n] = 0u;
  }
  for (int n = tid; n < NCN1; n += blockDim.x) {
    v_sa1[n] = -65.0f; u_sa1[n] = -13.0f; b_sa1[n] = 0u;
    v_sa2[n] = -65.0f; u_sa2[n] = -13.0f; b_sa2[n] = 0u;
    v_ra1[n] = -65.0f; u_ra1[n] = -13.0f; b_ra1[n] = 0u;
    v_ra2[n] = -65.0f; u_ra2[n] = -13.0f; b_ra2[n] = 0u;
    psp_sa[n] = 0.0f;  psp_ra[n] = 0.0f;
  }
  for (int n = tid; n < NCN; n += blockDim.x) {
    v_cn0[n] = -65.0f; u_cn0[n] = -13.0f; b_cn0[n] = 0u;
    v_cn1[n] = -65.0f; u_cn1[n] = -13.0f;
    psp_cn[n] = 0.0f;
  }
  __syncthreads();

  const float* I_sa_ff = I_ff;
  const float* I_ra_ff = I_ff + (size_t)T_STEPS * NSA;
  float* out_sa = out;
  float* out_ra = out + (size_t)T_STEPS * NSA;

  for (int t = 0; t < T_STEPS; ++t) {
    // -- 1: sensory layers (feed-forward currents from GEMM) --
    for (int n = tid; n < NSA; n += blockDim.x) {
      float s0 = izh_step(v_sa0[n], u_sa0[n], I_sa_ff[(size_t)t * NSA + n],
                          0.02f, 0.2f, -65.0f, 8.0f);
      b_sa0[n] = (b_sa0[n] << 1) | (unsigned)s0;
      out_sa[(size_t)t * NSA + n] = s0;

      float s1 = izh_step(v_ra0[n], u_ra0[n], I_ra_ff[(size_t)t * NSA + n],
                          0.02f, 0.2f, -65.0f, 2.0f);
      b_ra0[n] = (b_ra0[n] << 1) | (unsigned)s1;
      out_ra[(size_t)t * NSA + n] = s1;
    }
    __syncthreads();

    // -- 2: delayed synapses sa0->{sa1,pn}, ra0->{ra1,rpn} --
    syn_dual(sa_cn_in_w, sa_cn_pn_w, sa_cn_SD, b_sa0, NCN1, NSA, I_sa1, pn1, wave, lane, nwav);
    syn_dual(ra_cn_in_w, ra_cn_pn_w, ra_cn_SD, b_ra0, NCN1, NSA, I_ra1, rpn1, wave, lane, nwav);
    __syncthreads();

    // -- 3: interneurons sa1 / ra1 --
    for (int n = tid; n < NCN1; n += blockDim.x) {
      float s = izh_step(v_sa1[n], u_sa1[n], I_sa1[n], 0.1f, 0.2f, -65.0f, 6.0f);
      b_sa1[n] = (b_sa1[n] << 1) | (unsigned)s;
      float r = izh_step(v_ra1[n], u_ra1[n], I_ra1[n], 0.1f, 0.2f, -65.0f, 2.0f);
      b_ra1[n] = (b_ra1[n] << 1) | (unsigned)r;
    }
    __syncthreads();

    // -- 4: inhibitory PSP contributions (reuse I_sa1 / I_ra1 as scratch) --
    syn_dual(sa_intopn_w, nullptr, sa_intopn_DN, b_sa1, NCN1, NCN1, I_sa1, nullptr, wave, lane, nwav);
    syn_dual(ra_intopn_w, nullptr, ra_intopn_DN, b_ra1, NCN1, NCN1, I_ra1, nullptr, wave, lane, nwav);
    __syncthreads();

    // -- 5: projection neurons sa2 / ra2 --
    for (int n = tid; n < NCN1; n += blockDim.x) {
      psp_sa[n] = psp_sa[n] * PSPDEC + I_sa1[n];
      float s = izh_step(v_sa2[n], u_sa2[n], pn1[n] * 2.0f - psp_sa[n],
                         0.1f, 0.2f, -65.0f, 6.0f);
      b_sa2[n] = (b_sa2[n] << 1) | (unsigned)s;

      psp_ra[n] = psp_ra[n] * PSPDEC + I_ra1[n];
      float r = izh_step(v_ra2[n], u_ra2[n], rpn1[n] * 2.0f - psp_ra[n],
                         0.1f, 0.2f, -65.0f, 2.0f);
      b_ra2[n] = (b_ra2[n] << 1) | (unsigned)r;
    }
    __syncthreads();

    // -- 6: cortical inputs from sa2/ra2 --
    syn_dual(cn_in_sa_w, cn_pn_sa_w, cn_sa_SD, b_sa2, NCN, NCN1, cnA, cnB, wave, lane, nwav);
    syn_dual(cn_in_ra_w, cn_pn_ra_w, cn_ra_SD, b_ra2, NCN, NCN1, cnC, cnD, wave, lane, nwav);
    __syncthreads();

    // -- 7: cn0 --
    for (int n = tid; n < NCN; n += blockDim.x) {
      float s = izh_step(v_cn0[n], u_cn0[n], cnA[n] + cnC[n], 0.02f, 0.2f, -65.0f, 8.0f);
      b_cn0[n] = (b_cn0[n] << 1) | (unsigned)s;
    }
    __syncthreads();

    // -- 8: cn interneuron PSP (reuse cnA) --
    syn_dual(cn_intopn_w, nullptr, cn_intopn_DN, b_cn0, NCN, NCN, cnA, nullptr, wave, lane, nwav);
    __syncthreads();

    // -- 9: cn1 --
    for (int n = tid; n < NCN; n += blockDim.x) {
      psp_cn[n] = psp_cn[n] * PSPDEC + cnA[n];
      float I = cnB[n] * 2.0f + cnD[n] * 2.0f - psp_cn[n];
      (void)izh_step(v_cn1[n], u_cn1[n], I, 0.02f, 0.2f, -65.0f, 8.0f);
    }
    __syncthreads();
  }
}

extern "C" void kernel_launch(void* const* d_in, const int* in_sizes, int n_in,
                              void* d_out, int out_size, void* d_ws, size_t ws_size,
                              hipStream_t stream) {
  (void)in_sizes; (void)n_in; (void)out_size; (void)ws_size;

  const float* stim        = (const float*)d_in[0];
  const float* diff_stim   = (const float*)d_in[1];
  const float* sa_rf       = (const float*)d_in[2];
  const float* ra_rf       = (const float*)d_in[3];
  const float* sa_cn_in_w  = (const float*)d_in[4];
  const float* sa_cn_pn_w  = (const float*)d_in[5];
  const float* sa_intopn_w = (const float*)d_in[6];
  const float* ra_cn_in_w  = (const float*)d_in[7];
  const float* ra_cn_pn_w  = (const float*)d_in[8];
  const float* ra_intopn_w = (const float*)d_in[9];
  const float* cn_in_sa_w  = (const float*)d_in[10];
  const float* cn_pn_sa_w  = (const float*)d_in[11];
  const float* cn_in_ra_w  = (const float*)d_in[12];
  const float* cn_pn_ra_w  = (const float*)d_in[13];
  const float* cn_intopn_w = (const float*)d_in[14];
  const int*   sa_cn_SD    = (const int*)d_in[15];
  const int*   sa_intopn_DN= (const int*)d_in[16];
  const int*   ra_cn_SD    = (const int*)d_in[17];
  const int*   ra_intopn_DN= (const int*)d_in[18];
  const int*   cn_sa_SD    = (const int*)d_in[19];
  const int*   cn_ra_SD    = (const int*)d_in[20];
  const int*   cn_intopn_DN= (const int*)d_in[21];

  float* part = (float*)d_ws;   // [KSPLIT][2][T][NSA] = 7.68 MB partials

  // Phase A: KSPLIT*2*MT = 1200 waves, 8 waves/block -> 150 blocks
  snn_rf_gemm<<<150, 256, 0, stream>>>(stim, diff_stim, sa_rf, ra_rf, part);

  // Phase A2: reduce partials into slab 0
  snn_reduce<<<(SLAB + 255) / 256, 256, 0, stream>>>(part);

  // Phase B: one workgroup, full scan (reads slab 0)
  snn_scan<<<1, 1024, 0, stream>>>(
      part,
      sa_cn_in_w, sa_cn_pn_w, sa_intopn_w,
      ra_cn_in_w, ra_cn_pn_w, ra_intopn_w,
      cn_in_sa_w, cn_pn_sa_w, cn_in_ra_w, cn_pn_ra_w, cn_intopn_w,
      sa_cn_SD, sa_intopn_DN, ra_cn_SD, ra_intopn_DN,
      cn_sa_SD, cn_ra_SD, cn_intopn_DN,
      (float*)d_out);
}
